// SlotAttention_73701638799405
// MI455X (gfx1250) — compile-verified, matrix operand order checked by reading, not run on hardware
//
#include <hip/hip_runtime.h>
#include <hip/hip_bf16.h>

#define B_ 32
#define N_ 4096
#define C_ 256
#define M_ 7
#define D_ 128
#define H_ 256
#define KS_ 8          // K-splits for the updates GEMM
#define EPS_ 1e-6f
#define LN_EPS_ 1e-5f

typedef __attribute__((ext_vector_type(16))) _Float16 v16h;
typedef __attribute__((ext_vector_type(8)))  _Float16 v8h;
typedef __attribute__((ext_vector_type(8)))  float    v8f;
typedef __attribute__((ext_vector_type(4)))  unsigned int v4u;
typedef __attribute__((ext_vector_type(8)))  unsigned int v8u;

// ---------- helpers ----------

static __device__ inline float wsum(float v) {
#pragma unroll
  for (int m = 16; m >= 1; m >>= 1) v += __shfl_xor(v, m, 32);
  return v;
}

// 16-bit A-matrix 16x32 fragment from row-major memory (ld in halves).
static __device__ inline v16h load_a16(const _Float16* __restrict__ p, int ld) {
  int lane = threadIdx.x & 31;
  int m = lane & 15;
  int k0 = (lane & 16) ? 8 : 0;
  const _Float16* r = p + (size_t)m * ld + k0;
  v8h lo = *(const v8h*)r;
  v8h hi = *(const v8h*)(r + 16);
  v16h a;
#pragma unroll
  for (int i = 0; i < 8; ++i) { a[i] = lo[i]; a[i + 8] = hi[i]; }
  return a;
}

// B fragment from pre-swizzled tile: 512 halves, lane-major (lane*16 + h).
static __device__ inline v16h load_b_sw(const _Float16* __restrict__ t) {
  int lane = threadIdx.x & 31;
  const _Float16* r = t + lane * 16;
  v8h lo = *(const v8h*)r;
  v8h hi = *(const v8h*)(r + 8);
  v16h b;
#pragma unroll
  for (int i = 0; i < 8; ++i) { b[i] = lo[i]; b[i + 8] = hi[i]; }
  return b;
}

// B fragment (32 K-rows x 16 cols) from row-major [K][n] with leading dim ld.
static __device__ inline v16h load_b_rm(const _Float16* __restrict__ p, int ld) {
  int lane = threadIdx.x & 31;
  int n = lane & 15;
  int k0 = (lane & 16) ? 8 : 0;
  v16h b;
#pragma unroll
  for (int i = 0; i < 8; ++i) {
    b[i]     = p[(k0 + i) * ld + n];
    b[i + 8] = p[(k0 + 16 + i) * ld + n];
  }
  return b;
}

static __device__ inline v8f wmma16(v16h a, v16h b, v8f c) {
  return __builtin_amdgcn_wmma_f32_16x16x32_f16(false, a, false, b, (short)0, c,
                                                false, false);
}

// TDM: async-load one 32-row x 128-col f16 tile (row stride 128 halves in
// memory) into LDS at byte offset ldsOff, padding +4 DWORDs after each
// 64-DWORD row so the LDS image has a 136-half row stride.
// D# per CDNA5 ISA.
static __device__ inline void tdm_load_tile(const _Float16* gsrc, unsigned ldsOff) {
  unsigned long long ga = (unsigned long long)(uintptr_t)gsrc;
  v4u g0;
  g0[0] = 1u;                                      // count=1, user mode
  g0[1] = ldsOff;                                  // lds_addr
  g0[2] = (unsigned)ga;                            // global_addr[31:0]
  g0[3] = (unsigned)(ga >> 32) | 0x80000000u;      // addr[56:32] | type=2<<30
  v8u g1;
  g1[0] = (1u << 16)        // data_size = 2 bytes
        | (1u << 20)        // pad_enable
        | (5u << 22)        // pad_interval: 64 DWORDs
        | (3u << 25);       // pad_amount: 4 DWORDs
  g1[1] = (128u << 16);     // tensor_dim0 = 128
  g1[2] = (32u << 16);      // tensor_dim1 = 32
  g1[3] = (128u << 16);     // tile_dim0 = 128
  g1[4] = 32u;              // tile_dim1 = 32, tile_dim2 = 0
  g1[5] = 128u;             // tensor_dim0_stride = 128
  g1[6] = 0u;
  g1[7] = 0u;
  asm volatile("tensor_load_to_lds %0, %1" :: "s"(g0), "s"(g1) : "memory");
}

// ---------- kernels ----------

// Swizzle Wk/Wv ([Dout=128][Cin=256] fp32) into B-fragment layout.
__global__ void __launch_bounds__(512) prep_weights(
    const float* __restrict__ Wk, const float* __restrict__ Wv,
    _Float16* __restrict__ WkSw, _Float16* __restrict__ WvSw) {
  int bi = blockIdx.x;                       // 0..127
  const float* W = (bi < 64) ? Wk : Wv;
  _Float16* Wsw  = (bi < 64) ? WkSw : WvSw;
  int tile = bi & 63;                        // colTile*8 + kb
  int colTile = tile >> 3;
  int kb = tile & 7;
  int t = threadIdx.x;
  int lane = t >> 4;
  int h = t & 15;
  int k = kb * 32 + ((lane & 16) ? 8 : 0) + (h & 7) + ((h & 8) ? 16 : 0);
  int n = colTile * 16 + (lane & 15);
  Wsw[tile * 512 + t] = (_Float16)W[(size_t)n * C_ + k];
}

__global__ void copy_f32(const float* __restrict__ src, float* __restrict__ dst, int n) {
  int i = blockIdx.x * blockDim.x + threadIdx.x;
  if (i < n) dst[i] = src[i];
}

// Fused LayerNorm(inputs) + K/V projection via WMMA (16 rows / block, 16 waves).
__global__ void __launch_bounds__(512) ln_proj_kv(
    const float* __restrict__ x, const float* __restrict__ lnw,
    const float* __restrict__ lnb, const _Float16* __restrict__ WkSw,
    const _Float16* __restrict__ WvSw, _Float16* __restrict__ kf,
    _Float16* __restrict__ vf) {
  __shared__ _Float16 xln[16][272];
  int row0 = blockIdx.x * 16;
  int wave = threadIdx.x >> 5;
  int lane = threadIdx.x & 31;
  {  // LN: one wave per row
    const float* xr = x + (size_t)(row0 + wave) * C_;
    float vals[8]; float s = 0.f, sq = 0.f;
#pragma unroll
    for (int j = 0; j < 8; ++j) {
      float xv = xr[lane + j * 32];
      vals[j] = xv; s += xv; sq += xv * xv;
    }
    s = wsum(s); sq = wsum(sq);
    float mean = s * (1.0f / C_);
    float var = sq * (1.0f / C_) - mean * mean;
    float rstd = rsqrtf(var + LN_EPS_);
#pragma unroll
    for (int j = 0; j < 8; ++j) {
      int c = lane + j * 32;
      xln[wave][c] = (_Float16)((vals[j] - mean) * rstd * lnw[c] + lnb[c]);
    }
  }
  __syncthreads();
  int colTile = wave & 7;
  const _Float16* Wsw = (wave < 8) ? WkSw : WvSw;
  _Float16* out = (wave < 8) ? kf : vf;
  v8f c = {};
#pragma unroll
  for (int kb = 0; kb < 8; ++kb) {
    v16h a = load_a16(&xln[0][0] + kb * 32, 272);
    v16h b = load_b_sw(Wsw + (colTile * 8 + kb) * 512);
    c = wmma16(a, b, c);
  }
  int col = colTile * 16 + (lane & 15);
  int rb = (lane & 16) ? 8 : 0;
#pragma unroll
  for (int r = 0; r < 8; ++r)
    out[(size_t)(row0 + rb + r) * D_ + col] = (_Float16)c[r];
}

// q = LN(slots) @ Wq.T, scattered to swizzled B-fragment layout [B][4 kb][512].
__global__ void __launch_bounds__(128) q_proj(
    const float* __restrict__ slots, const float* __restrict__ lw,
    const float* __restrict__ lb, const float* __restrict__ Wq,
    _Float16* __restrict__ qsw) {
  __shared__ float sy[D_];
  __shared__ float sred[4];
  int b = blockIdx.x / M_;
  int m = blockIdx.x % M_;
  int t = threadIdx.x;
  float sv = slots[(size_t)(b * M_ + m) * D_ + t];
  float s = wsum(sv);
  if ((t & 31) == 0) sred[t >> 5] = s;
  __syncthreads();
  float mean = (sred[0] + sred[1] + sred[2] + sred[3]) * (1.0f / D_);
  __syncthreads();
  float d = sv - mean;
  float q2 = wsum(d * d);
  if ((t & 31) == 0) sred[t >> 5] = q2;
  __syncthreads();
  float var = (sred[0] + sred[1] + sred[2] + sred[3]) * (1.0f / D_);
  sy[t] = d * rsqrtf(var + LN_EPS_) * lw[t] + lb[t];
  __syncthreads();
  float acc = 0.f;
  const float* wr = Wq + (size_t)t * D_;
  for (int j = 0; j < D_; ++j) acc += wr[j] * sy[j];
  int kOff = t & 31, kb = t >> 5;
  int hi = (kOff >> 3) & 1;
  int h = (kOff & 7) + ((kOff & 16) ? 8 : 0);
  int lane = m + hi * 16;
  qsw[(size_t)b * 2048 + kb * 512 + lane * 16 + h] = (_Float16)acc;
}

// logits = scale * k @ q^T via WMMA, fused softmax over slots, write attn^T f16.
__global__ void __launch_bounds__(256) logits_softmax(
    const _Float16* __restrict__ kf, const _Float16* __restrict__ qsw,
    _Float16* __restrict__ attn) {
  __shared__ float lg[8][16][17];
  int b = blockIdx.y;
  int nBase = blockIdx.x * 128;
  int wave = threadIdx.x >> 5;
  int lane = threadIdx.x & 31;
  int n0 = nBase + wave * 16;
  const _Float16* ka = kf + (size_t)(b * N_ + n0) * D_;
  v8f c = {};
#pragma unroll
  for (int kb = 0; kb < 4; ++kb) {
    v16h a = load_a16(ka + kb * 32, D_);
    v16h bq = load_b_sw(qsw + (size_t)b * 2048 + kb * 512);
    c = wmma16(a, bq, c);
  }
  const float scale = 0.08838834764831845f;  // 1/sqrt(D)
  int col = lane & 15;
  int rb = (lane & 16) ? 8 : 0;
#pragma unroll
  for (int r = 0; r < 8; ++r) lg[wave][rb + r][col] = c[r] * scale;
  __syncthreads();
  int t = threadIdx.x;
  if (t < 128) {
    int tile = t >> 4, nl = t & 15;
    float l[M_], mx = -1e30f;
#pragma unroll
    for (int m = 0; m < M_; ++m) { l[m] = lg[tile][nl][m]; mx = fmaxf(mx, l[m]); }
    float sum = 0.f;
#pragma unroll
    for (int m = 0; m < M_; ++m) { l[m] = __expf(l[m] - mx); sum += l[m]; }
    float inv = 1.0f / sum;
    int n = nBase + tile * 16 + nl;
#pragma unroll
    for (int m = 0; m < M_; ++m)
      attn[((size_t)(b * 16 + m)) * N_ + n] = (_Float16)(l[m] * inv + EPS_);
  }
}

// Deterministic per-(b,m) sum over N; store reciprocal.
__global__ void __launch_bounds__(256) attn_rowsum(
    const _Float16* __restrict__ attn, float* __restrict__ rrec) {
  int b = blockIdx.x / M_, m = blockIdx.x % M_;
  const _Float16* row = attn + ((size_t)(b * 16 + m)) * N_;
  float s = 0.f;
  for (int j = threadIdx.x; j < N_; j += 256) s += (float)row[j];
  __shared__ float sred[8];
  float ws = wsum(s);
  if ((threadIdx.x & 31) == 0) sred[threadIdx.x >> 5] = ws;
  __syncthreads();
  if (threadIdx.x == 0) {
    float tot = 0.f;
#pragma unroll
    for (int i = 0; i < 8; ++i) tot += sred[i];
    rrec[b * 16 + m] = 1.0f / tot;
  }
}

// updates partials: (attn^T) @ v over a K-slice of 512 rows, staged via TDM
// double-buffered LDS tiles. Grid (KS_, B), 256 threads = 8 waves.
__global__ void __launch_bounds__(256) attn_updates(
    const _Float16* __restrict__ attn, const _Float16* __restrict__ vf,
    float* __restrict__ updp) {
  __shared__ _Float16 vt[2][32 * 136];   // only LDS object -> base offset 0
  int ks = blockIdx.x;
  int b = blockIdx.y;
  int wave = threadIdx.x >> 5;
  int lane = threadIdx.x & 31;
  const int chunks = (N_ / KS_) / 32;    // 16
  int row0 = ks * (N_ / KS_);
  const _Float16* attnB = attn + (size_t)b * 16 * N_ + row0;
  const _Float16* vbase = vf + (size_t)(b * N_ + row0) * D_;

  if (wave == 0) tdm_load_tile(vbase, 0u);
  v8f c = {};
  for (int kk = 0; kk < chunks; ++kk) {
    if (wave == 0) {
      if (kk + 1 < chunks) {
        tdm_load_tile(vbase + (size_t)(kk + 1) * 32 * D_,
                      (unsigned)(((kk + 1) & 1) * 32 * 136 * 2));
        __builtin_amdgcn_s_wait_tensorcnt(1);
      } else {
        __builtin_amdgcn_s_wait_tensorcnt(0);
      }
    }
    __syncthreads();                      // tile kk ready for all waves
    v16h a = load_a16(attnB + kk * 32, N_);
    v16h bv = load_b_rm(&vt[kk & 1][0] + wave * 16, 136);
    c = wmma16(a, bv, c);
    __syncthreads();                      // reads done before buffer reuse
  }
  int col = wave * 16 + (lane & 15);
  int rb = (lane & 16) ? 8 : 0;
#pragma unroll
  for (int r = 0; r < 8; ++r)
    updp[(((size_t)(b * KS_ + ks) * 16) + rb + r) * D_ + col] = c[r];
}

// Deterministic fixed-order reduction of K-split partials + 1/colsum scaling.
__global__ void __launch_bounds__(256) upd_reduce(
    const float* __restrict__ updp, const float* __restrict__ rrec,
    float* __restrict__ upd) {
  int i = blockIdx.x * 256 + threadIdx.x;  // over B*16*D
  int d = i & (D_ - 1);
  int bm = i >> 7;                         // b*16+m
  int b = bm >> 4;
  float s = 0.f;
#pragma unroll
  for (int ks = 0; ks < KS_; ++ks)
    s += updp[(((size_t)(b * KS_ + ks) * 16) + (bm & 15)) * D_ + d];
  upd[(size_t)bm * D_ + d] = s * rrec[bm];
}

// Fused GRUCell + residual MLP per slot row (224 rows; scalar fp32).
__global__ void __launch_bounds__(128) gru_mlp(
    const float* __restrict__ upd, float* __restrict__ slots,
    const float* __restrict__ gwi, const float* __restrict__ gwh,
    const float* __restrict__ gbi, const float* __restrict__ gbh,
    const float* __restrict__ lmw, const float* __restrict__ lmb,
    const float* __restrict__ w1, const float* __restrict__ b1,
    const float* __restrict__ w2, const float* __restrict__ b2) {
  __shared__ float su[D_], sh[D_], sy[D_], sh1[H_];
  __shared__ float sred[4];
  int b = blockIdx.x / M_, m = blockIdx.x % M_;
  int t = threadIdx.x;
  su[t] = upd[((size_t)(b * 16 + m)) * D_ + t];
  float hv = slots[(size_t)(b * M_ + m) * D_ + t];
  sh[t] = hv;
  __syncthreads();
  float ir = gbi[t], iz = gbi[D_ + t], in_ = gbi[2 * D_ + t];
  float hr = gbh[t], hz = gbh[D_ + t], hn = gbh[2 * D_ + t];
  for (int j = 0; j < D_; ++j) {
    float uj = su[j], hj = sh[j];
    ir += gwi[(size_t)t * D_ + j] * uj;
    iz += gwi[(size_t)(D_ + t) * D_ + j] * uj;
    in_ += gwi[(size_t)(2 * D_ + t) * D_ + j] * uj;
    hr += gwh[(size_t)t * D_ + j] * hj;
    hz += gwh[(size_t)(D_ + t) * D_ + j] * hj;
    hn += gwh[(size_t)(2 * D_ + t) * D_ + j] * hj;
  }
  float r = 1.0f / (1.0f + __expf(-(ir + hr)));
  float z = 1.0f / (1.0f + __expf(-(iz + hz)));
  float n = tanhf(in_ + r * hn);
  float s = (1.0f - z) * n + z * hv;
  __syncthreads();
  float sm = wsum(s);
  if ((t & 31) == 0) sred[t >> 5] = sm;
  __syncthreads();
  float mean = (sred[0] + sred[1] + sred[2] + sred[3]) * (1.0f / D_);
  __syncthreads();
  float d = s - mean;
  float vq = wsum(d * d);
  if ((t & 31) == 0) sred[t >> 5] = vq;
  __syncthreads();
  float var = (sred[0] + sred[1] + sred[2] + sred[3]) * (1.0f / D_);
  sy[t] = d * rsqrtf(var + LN_EPS_) * lmw[t] + lmb[t];
  __syncthreads();
#pragma unroll
  for (int o = 0; o < 2; ++o) {
    int oo = t + o * D_;
    float a = b1[oo];
    for (int j = 0; j < D_; ++j) a += w1[(size_t)oo * D_ + j] * sy[j];
    sh1[oo] = fmaxf(a, 0.0f);
  }
  __syncthreads();
  float out = b2[t];
  for (int j = 0; j < H_; ++j) out += w2[(size_t)t * H_ + j] * sh1[j];
  slots[(size_t)(b * M_ + m) * D_ + t] = s + out;
}

// ---------- launcher ----------

extern "C" void kernel_launch(void* const* d_in, const int* in_sizes, int n_in,
                              void* d_out, int out_size, void* d_ws, size_t ws_size,
                              hipStream_t stream) {
  (void)in_sizes; (void)n_in; (void)out_size; (void)ws_size;
  const float* inputs    = (const float*)d_in[0];
  const float* slots_mu  = (const float*)d_in[1];
  const float* ln_in_w   = (const float*)d_in[2];
  const float* ln_in_b   = (const float*)d_in[3];
  const float* ln_slot_w = (const float*)d_in[4];
  const float* ln_slot_b = (const float*)d_in[5];
  const float* ln_mlp_w  = (const float*)d_in[6];
  const float* ln_mlp_b  = (const float*)d_in[7];
  const float* Wq        = (const float*)d_in[8];
  const float* Wk        = (const float*)d_in[9];
  const float* Wv        = (const float*)d_in[10];
  const float* gwi       = (const float*)d_in[11];
  const float* gwh       = (const float*)d_in[12];
  const float* gbi       = (const float*)d_in[13];
  const float* gbh       = (const float*)d_in[14];
  const float* w1        = (const float*)d_in[15];
  const float* b1        = (const float*)d_in[16];
  const float* w2        = (const float*)d_in[17];
  const float* b2        = (const float*)d_in[18];

  char* ws = (char*)d_ws;
  size_t off = 0;
  auto alloc = [&](size_t bytes) -> char* {
    char* p = ws + off;
    off = (off + bytes + 255) & ~(size_t)255;
    return p;
  };
  _Float16* kf    = (_Float16*)alloc((size_t)B_ * N_ * D_ * 2);
  _Float16* vf    = (_Float16*)alloc((size_t)B_ * N_ * D_ * 2);
  _Float16* attn  = (_Float16*)alloc((size_t)B_ * 16 * N_ * 2);
  _Float16* qsw   = (_Float16*)alloc((size_t)B_ * 2048 * 2);
  _Float16* WkSw  = (_Float16*)alloc((size_t)64 * 512 * 2);
  _Float16* WvSw  = (_Float16*)alloc((size_t)64 * 512 * 2);
  float*    rrec  = (float*)alloc((size_t)B_ * 16 * 4);
  float*    updp  = (float*)alloc((size_t)B_ * KS_ * 16 * D_ * 4);
  float*    upd   = (float*)alloc((size_t)B_ * 16 * D_ * 4);
  float*    slots = (float*)alloc((size_t)B_ * M_ * D_ * 4);

  prep_weights<<<128, 512, 0, stream>>>(Wk, Wv, WkSw, WvSw);
  copy_f32<<<(B_ * M_ * D_ + 255) / 256, 256, 0, stream>>>(slots_mu, slots, B_ * M_ * D_);
  ln_proj_kv<<<(B_ * N_) / 16, 512, 0, stream>>>(inputs, ln_in_w, ln_in_b, WkSw, WvSw, kf, vf);
  for (int it = 0; it < 3; ++it) {
    q_proj<<<B_ * M_, 128, 0, stream>>>(slots, ln_slot_w, ln_slot_b, Wq, qsw);
    logits_softmax<<<dim3(N_ / 128, B_), 256, 0, stream>>>(kf, qsw, attn);
    attn_rowsum<<<B_ * M_, 256, 0, stream>>>(attn, rrec);
    attn_updates<<<dim3(KS_, B_), 256, 0, stream>>>(attn, vf, updp);
    upd_reduce<<<(B_ * 16 * D_) / 256, 256, 0, stream>>>(updp, rrec, upd);
    gru_mlp<<<B_ * M_, 128, 0, stream>>>(upd, slots, gwi, gwh, gbi, gbh,
                                         ln_mlp_w, ln_mlp_b, w1, b1, w2, b2);
  }
  copy_f32<<<(B_ * M_ * D_ + 255) / 256, 256, 0, stream>>>(slots, (float*)d_out, B_ * M_ * D_);
}
